// DualAttention_11218454577558
// MI455X (gfx1250) — matrix operand order repaired and checked by reading, not compiled, not run on hardware
//
#include <hip/hip_runtime.h>

// ---------------------------------------------------------------------------
// DANet-style dual attention for MI455X (gfx1250, wave32, WMMA f16->f32).
// b=2, c=512, n=4096, d=64.  All GEMMs via v_wmma_f32_16x16x32_f16.
// Position attention is fused flash-style; the V stream is staged into LDS
// with CDNA5 async loads (GLOBAL_LOAD_ASYNC_TO_LDS_B128, ASYNCcnt).
// ---------------------------------------------------------------------------

typedef __attribute__((ext_vector_type(16))) _Float16 v16h;
typedef __attribute__((ext_vector_type(8)))  _Float16 v8h;
typedef __attribute__((ext_vector_type(8)))  float    v8f;

#define WMMA(A, B, C) \
  __builtin_amdgcn_wmma_f32_16x16x32_f16(false, (A), false, (B), (short)0, (C), false, false)

// A fragment (16x32 f16, M x K): lane l: m=l&15, h=l>>4.
// halves 0..7  <- k = 8h+0..7 ;  halves 8..15 <- k = 16+8h+0..7   (ISA 7.12.2)
__device__ __forceinline__ v16h load_a_frag(const _Float16* p, int ld) {
  const int l = threadIdx.x & 31, m = l & 15, h = l >> 4;
  const _Float16* r = p + m * ld + 8 * h;
  v8h lo = *(const v8h*)r;
  v8h hi = *(const v8h*)(r + 16);
  v16h a;
#pragma unroll
  for (int t = 0; t < 8; ++t) { a[t] = lo[t]; a[t + 8] = hi[t]; }
  return a;
}

// B fragment (32x16 f16, K x N), element(k,n) = p[n*ld + k]:
// lane l: n=l&15, h=l>>4; halves 0..7 <- k = 8h+0..7 ; halves 8..15 <- k = 16+8h+0..7
__device__ __forceinline__ v16h load_b_frag(const _Float16* p, int ld) {
  const int l = threadIdx.x & 31, n = l & 15, h = l >> 4;
  const _Float16* r = p + n * ld + 8 * h;
  v8h lo = *(const v8h*)r;
  v8h hi = *(const v8h*)(r + 16);
  v16h b;
#pragma unroll
  for (int t = 0; t < 8; ++t) { b[t] = lo[t]; b[t + 8] = hi[t]; }
  return b;
}

// Generic -> LDS byte offset (addrspacecast then truncate).
__device__ __forceinline__ unsigned int lds_offset_of(const void* p) {
  return (unsigned int)(unsigned long long)
      (const __attribute__((address_space(3))) void*)p;
}

// Issue one 32(j) x 512(c) f16 V block (32 KB) Global->LDS with async copies.
// 2048 16-byte chunks, 8 per lane. LDS row stride = 40 halves (80 B): 16-byte
// aligned and bank-conflict-free for the later per-row ds_load_b128 reads.
__device__ __forceinline__ void async_stage_v(const _Float16* __restrict__ vh,
                                              int j0, _Float16* dst,
                                              int w, int l) {
#pragma unroll
  for (int q = 0; q < 8; ++q) {
    const int chunk = (q * 8 + w) * 32 + l;   // 0..2047
    const int c  = chunk >> 2;                 // 0..511
    const int jq = chunk & 3;                  // 0..3 (8 halves each)
    const _Float16* g = vh + (size_t)c * 4096 + j0 + jq * 8;
    unsigned int loff = lds_offset_of(dst + c * 40 + jq * 8);
    asm volatile("global_load_async_to_lds_b128 %0, %1, off"
                 :: "v"(loff), "v"(g) : "memory");
  }
}

__device__ __forceinline__ void wait_async_all() {
  asm volatile("s_wait_asynccnt 0x0" ::: "memory");
}

// ---------------------------------------------------------------------------
// Cast x (b,c,n) f32 -> XH (b,c,n) f16 and XHT (b,n,c) f16 (LDS transpose).
// grid (n/32, c/32, b), block (32,8)
__global__ __launch_bounds__(256) void cast_x_kernel(
    const float* __restrict__ x, _Float16* __restrict__ XH, _Float16* __restrict__ XHT) {
  __shared__ _Float16 t[32][33];
  const int b = blockIdx.z, n0 = blockIdx.x * 32, c0 = blockIdx.y * 32;
  const size_t boff = (size_t)b * 512 * 4096;
#pragma unroll
  for (int k = 0; k < 4; ++k) {
    int c = c0 + threadIdx.y + k * 8;
    float v = x[boff + (size_t)c * 4096 + n0 + threadIdx.x];
    _Float16 hv = (_Float16)v;
    XH[boff + (size_t)c * 4096 + n0 + threadIdx.x] = hv;
    t[threadIdx.y + k * 8][threadIdx.x] = hv;
  }
  __syncthreads();
#pragma unroll
  for (int k = 0; k < 4; ++k) {
    int n = n0 + threadIdx.y + k * 8;
    XHT[boff + (size_t)n * 512 + c0 + threadIdx.x] = t[threadIdx.x][threadIdx.y + k * 8];
  }
}

// Cast weights to f16. grid(1024), block(256)
__global__ __launch_bounds__(256) void cast_w_kernel(
    const float* __restrict__ wq, const float* __restrict__ wk, const float* __restrict__ wv,
    _Float16* __restrict__ wqh, _Float16* __restrict__ wkh, _Float16* __restrict__ wvh) {
  int i = blockIdx.x * 256 + threadIdx.x;
  if (i < 64 * 512) { wqh[i] = (_Float16)wq[i]; wkh[i] = (_Float16)wk[i]; }
  wvh[i] = (_Float16)wv[i];  // 512*512 = grid exactly
}

// ---------------------------------------------------------------------------
// Projection GEMM: out(M,4096) = W(M,512) * X(512,4096) + bias.
// B element (k,n) read from XT[n*512+k] (both operands contiguous per lane).
// store_tr=1: out laid out (n, M) f16 (for Q^T / K^T);  else (M, n) f16 (V).
// grid (4096/64, M/32, b), block 256 (8 waves: 2 M-tiles x 4 N-tiles)
__global__ __launch_bounds__(256) void proj_gemm_kernel(
    const _Float16* __restrict__ W, const _Float16* __restrict__ XT,
    const float* __restrict__ bias, _Float16* __restrict__ out, int M, int store_tr) {
  const int w = threadIdx.x >> 5, l = threadIdx.x & 31, nl = l & 15, h = l >> 4;
  const int b = blockIdx.z;
  const int m0 = blockIdx.y * 32 + (w >> 2) * 16;
  const int n0 = blockIdx.x * 64 + (w & 3) * 16;
  const _Float16* xt = XT + (size_t)b * 4096 * 512;
  v8f acc = {0.f, 0.f, 0.f, 0.f, 0.f, 0.f, 0.f, 0.f};
#pragma unroll 4
  for (int k = 0; k < 512; k += 32) {
    v16h a  = load_a_frag(W + (size_t)m0 * 512 + k, 512);
    v16h bb = load_b_frag(xt + (size_t)n0 * 512 + k, 512);
    acc = WMMA(a, bb, acc);
  }
  float bs[8];
#pragma unroll
  for (int v = 0; v < 8; ++v) bs[v] = bias[m0 + 8 * h + v];
  if (store_tr) {
    v8h hv;
#pragma unroll
    for (int v = 0; v < 8; ++v) hv[v] = (_Float16)(acc[v] + bs[v]);
    *(v8h*)(out + (size_t)b * 4096 * M + (size_t)(n0 + nl) * M + m0 + 8 * h) = hv;
  } else {
#pragma unroll
    for (int v = 0; v < 8; ++v)
      out[(size_t)b * (size_t)M * 4096 + (size_t)(m0 + 8 * h + v) * 4096 + n0 + nl] =
          (_Float16)(acc[v] + bs[v]);
  }
}

// ---------------------------------------------------------------------------
// Channel energy E(512,512) = XH * XH^T (symmetric -> transposed stores are
// still correct and become contiguous v8f stores). grid (8, 4, 2), block 256
__global__ __launch_bounds__(256) void chan_energy_kernel(
    const _Float16* __restrict__ XH, float* __restrict__ EC) {
  const int w = threadIdx.x >> 5, l = threadIdx.x & 31, nl = l & 15, h = l >> 4;
  const int b = blockIdx.z;
  const int m0 = blockIdx.y * 128 + w * 16;
  const int n0 = blockIdx.x * 64;
  const _Float16* xb = XH + (size_t)b * 512 * 4096;
  v8f acc[4];
#pragma unroll
  for (int f = 0; f < 4; ++f) acc[f] = (v8f){0.f,0.f,0.f,0.f,0.f,0.f,0.f,0.f};
  for (int k = 0; k < 4096; k += 32) {
    v16h a = load_a_frag(xb + (size_t)m0 * 4096 + k, 4096);
#pragma unroll
    for (int f = 0; f < 4; ++f) {
      v16h bb = load_b_frag(xb + (size_t)(n0 + 16 * f) * 4096 + k, 4096);
      acc[f] = WMMA(a, bb, acc[f]);
    }
  }
  float* eb = EC + (size_t)b * 512 * 512;
#pragma unroll
  for (int f = 0; f < 4; ++f)
    *(v8f*)(eb + (size_t)(n0 + 16 * f + nl) * 512 + m0 + 8 * h) = acc[f];  // E^T == E
}

// Channel softmax: att = softmax(rowmax - e) == exp(rowmin - e)/sum.
// One block per row. grid(1024), block 256
__global__ __launch_bounds__(256) void chan_softmax_kernel(
    const float* __restrict__ EC, _Float16* __restrict__ ATTC) {
  __shared__ float red[256];
  const int r = blockIdx.x, b = r >> 9, i = r & 511, t = threadIdx.x;
  const float* e = EC + (size_t)b * 512 * 512 + (size_t)i * 512;
  float e0 = e[t], e1 = e[t + 256];
  red[t] = fminf(e0, e1);
  __syncthreads();
  for (int s = 128; s > 0; s >>= 1) { if (t < s) red[t] = fminf(red[t], red[t + s]); __syncthreads(); }
  float mn = red[0];
  __syncthreads();
  float p0 = __expf(mn - e0), p1 = __expf(mn - e1);
  red[t] = p0 + p1;
  __syncthreads();
  for (int s = 128; s > 0; s >>= 1) { if (t < s) red[t] += red[t + s]; __syncthreads(); }
  float inv = 1.0f / red[0];
  _Float16* a = ATTC + (size_t)b * 512 * 512 + (size_t)i * 512;
  a[t] = (_Float16)(p0 * inv);
  a[t + 256] = (_Float16)(p1 * inv);
}

// ---------------------------------------------------------------------------
// Flash position attention. Computes S^T = K*Q^T so the exp'd tile is already
// in WMMA A-fragment layout for P*V (no LDS transpose). Online softmax.
// V j-blocks are staged Global->LDS with async copies, double-buffered and
// issued one block ahead; shared by all 8 waves of the workgroup.
// grid (4096/32, b), block 256: wave = (i-subtile[2]) x (c-quarter[4]).
// Dynamic LDS: 2 * 512 rows * 40 halves * 2 B = 80 KB.
#define VROW 40            // padded LDS row stride in halves (80 B)
#define VBUF (512 * VROW)  // halves per buffer
__global__ __launch_bounds__(256) void pos_flash_kernel(
    const _Float16* __restrict__ QT, const _Float16* __restrict__ KT,
    const _Float16* __restrict__ VH, float* __restrict__ POS) {
  extern __shared__ _Float16 vtile[];
  const int w = threadIdx.x >> 5, l = threadIdx.x & 31, nl = l & 15, h = l >> 4;
  const int b = blockIdx.y;
  const int i0 = blockIdx.x * 32 + (w & 1) * 16;
  const int c0 = (w >> 1) * 128;
  const _Float16* qt = QT + (size_t)b * 4096 * 64;
  const _Float16* kt = KT + (size_t)b * 4096 * 64;
  const _Float16* vh = VH + (size_t)b * 512 * 4096;
  const v8f zf = {0.f,0.f,0.f,0.f,0.f,0.f,0.f,0.f};

  // Prologue: stage first V block into buffer 0.
  async_stage_v(vh, 0, vtile, w, l);

  // Q B-fragments for this i-tile (d = 0..31 and 32..63), resident all kernel.
  v16h bq0 = load_b_frag(qt + (size_t)i0 * 64, 64);
  v16h bq1 = load_b_frag(qt + (size_t)i0 * 64 + 32, 64);

  float mrow = -3.0e38f, lrow = 0.0f;  // stats for i = i0 + (l&15)
  v8f acc[8];
#pragma unroll
  for (int f = 0; f < 8; ++f) acc[f] = zf;

  wait_async_all();
  __syncthreads();

  for (int t = 0; t < 128; ++t) {
    const int j0 = t * 32;
    const _Float16* vbuf = vtile + (t & 1) * VBUF;
    if (t + 1 < 128)  // prefetch next V block into the other buffer
      async_stage_v(vh, j0 + 32, vtile + ((t + 1) & 1) * VBUF, w, l);
    if (j0 + 64 < 4096)  // L2 prefetch of the K stream two blocks ahead
      __builtin_prefetch(kt + (size_t)(j0 + 64) * 64, 0, 3);

    // S^T tiles: rows j, cols i.
    v8f st0 = WMMA(load_a_frag(kt + (size_t)j0 * 64, 64), bq0, zf);
    st0 = WMMA(load_a_frag(kt + (size_t)j0 * 64 + 32, 64), bq1, st0);
    v8f st1 = WMMA(load_a_frag(kt + (size_t)(j0 + 16) * 64, 64), bq0, zf);
    st1 = WMMA(load_a_frag(kt + (size_t)(j0 + 16) * 64 + 32, 64), bq1, st1);

    // Row (over j) max for this block: in-lane over v, then across halves.
    float bm = fmaxf(st0[0], st1[0]);
#pragma unroll
    for (int v = 1; v < 8; ++v) bm = fmaxf(bm, fmaxf(st0[v], st1[v]));
    bm = fmaxf(bm, __shfl_xor(bm, 16, 32));
    float mnew = fmaxf(mrow, bm);
    float sc = __expf(mrow - mnew);

    float p0[8], p1[8], rs = 0.0f;
#pragma unroll
    for (int v = 0; v < 8; ++v) {
      p0[v] = __expf(st0[v] - mnew);
      p1[v] = __expf(st1[v] - mnew);
      rs += p0[v] + p1[v];
    }
    rs += __shfl_xor(rs, 16, 32);
    lrow = lrow * sc + rs;
    mrow = mnew;

    // P tile already in A-fragment layout: halves 0..7 = j(8h+v), 8..15 = j(16+8h+v)
    v16h pa;
#pragma unroll
    for (int v = 0; v < 8; ++v) { pa[v] = (_Float16)p0[v]; pa[8 + v] = (_Float16)p1[v]; }

    // Broadcast per-row rescale factors to accumulator rows i = 8h+v.
    float scv[8];
#pragma unroll
    for (int v = 0; v < 8; ++v) scv[v] = __shfl(sc, 8 * h + v, 32);

#pragma unroll
    for (int f = 0; f < 8; ++f) {
#pragma unroll
      for (int v = 0; v < 8; ++v) acc[f][v] *= scv[v];
      v16h vb = load_b_frag(vbuf + (c0 + 16 * f) * VROW, VROW);  // ds_load_b128 x2
      acc[f] = WMMA(pa, vb, acc[f]);
    }

    // Next buffer must be complete (own waits) and all waves done reading
    // the current one before it is overwritten next iteration.
    wait_async_all();
    __syncthreads();
  }

  float linv[8];
#pragma unroll
  for (int v = 0; v < 8; ++v) linv[v] = 1.0f / __shfl(lrow, 8 * h + v, 32);
  float* pb = POS + (size_t)b * 512 * 4096;
#pragma unroll
  for (int f = 0; f < 8; ++f) {
    v8f o;
#pragma unroll
    for (int v = 0; v < 8; ++v) o[v] = acc[f][v] * linv[v];
    *(v8f*)(pb + (size_t)(c0 + 16 * f + nl) * 4096 + i0 + 8 * h) = o;
  }
}

// ---------------------------------------------------------------------------
// Channel out GEMM (att_c @ X) fused with final combine:
// out = alpha*gp*pos + beta*gc*chan + (alpha+beta+1)*x
// grid (64, 16, 2), block 256
__global__ __launch_bounds__(256) void chan_out_combine_kernel(
    const _Float16* __restrict__ ATTC, const _Float16* __restrict__ XHT,
    const float* __restrict__ POS, const float* __restrict__ x,
    const float* __restrict__ gp, const float* __restrict__ gc,
    const float* __restrict__ al, const float* __restrict__ be,
    float* __restrict__ out) {
  const int w = threadIdx.x >> 5, l = threadIdx.x & 31, nl = l & 15, h = l >> 4;
  const int b = blockIdx.z;
  const int m0 = blockIdx.y * 32 + (w >> 2) * 16;
  const int n0 = blockIdx.x * 64 + (w & 3) * 16;
  const _Float16* at = ATTC + (size_t)b * 512 * 512;
  const _Float16* xt = XHT + (size_t)b * 4096 * 512;
  v8f acc = {0.f, 0.f, 0.f, 0.f, 0.f, 0.f, 0.f, 0.f};
#pragma unroll 4
  for (int k = 0; k < 512; k += 32) {
    v16h a  = load_a_frag(at + (size_t)m0 * 512 + k, 512);
    v16h bb = load_b_frag(xt + (size_t)n0 * 512 + k, 512);
    acc = WMMA(a, bb, acc);
  }
  const float A = al[0] * gp[0], B = be[0] * gc[0], Cc = al[0] + be[0] + 1.0f;
  const size_t boff = (size_t)b * 512 * 4096;
  const int n = n0 + nl;
#pragma unroll
  for (int v = 0; v < 8; ++v) {
    size_t idx = boff + (size_t)(m0 + 8 * h + v) * 4096 + n;
    out[idx] = A * POS[idx] + B * acc[v] + Cc * x[idx];
  }
}

// ---------------------------------------------------------------------------
extern "C" void kernel_launch(void* const* d_in, const int* in_sizes, int n_in,
                              void* d_out, int out_size, void* d_ws, size_t ws_size,
                              hipStream_t stream) {
  (void)in_sizes; (void)n_in; (void)out_size; (void)ws_size;
  const float* x  = (const float*)d_in[0];
  const float* wq = (const float*)d_in[1];
  const float* bq = (const float*)d_in[2];
  const float* wk = (const float*)d_in[3];
  const float* bk = (const float*)d_in[4];
  const float* wv = (const float*)d_in[5];
  const float* bv = (const float*)d_in[6];
  const float* gp = (const float*)d_in[7];
  const float* gc = (const float*)d_in[8];
  const float* al = (const float*)d_in[9];
  const float* be = (const float*)d_in[10];
  float* out = (float*)d_out;

  char* ws = (char*)d_ws;
  size_t off = 0;
  auto take = [&](size_t bytes) { char* p = ws + off; off += (bytes + 255) & ~(size_t)255; return p; };
  _Float16* XH   = (_Float16*)take((size_t)2 * 512 * 4096 * 2);   // 8 MB
  _Float16* XHT  = (_Float16*)take((size_t)2 * 4096 * 512 * 2);   // 8 MB
  _Float16* WQH  = (_Float16*)take((size_t)64 * 512 * 2);
  _Float16* WKH  = (_Float16*)take((size_t)64 * 512 * 2);
  _Float16* WVH  = (_Float16*)take((size_t)512 * 512 * 2);
  _Float16* QT   = (_Float16*)take((size_t)2 * 4096 * 64 * 2);    // 1 MB
  _Float16* KT   = (_Float16*)take((size_t)2 * 4096 * 64 * 2);    // 1 MB
  _Float16* VH   = (_Float16*)take((size_t)2 * 512 * 4096 * 2);   // 8 MB
  float*    EC   = (float*)take((size_t)2 * 512 * 512 * 4);       // 2 MB
  _Float16* ATTC = (_Float16*)take((size_t)2 * 512 * 512 * 2);    // 1 MB
  float*    POS  = (float*)take((size_t)2 * 512 * 4096 * 4);      // 16 MB

  cast_x_kernel<<<dim3(128, 16, 2), dim3(32, 8), 0, stream>>>(x, XH, XHT);
  cast_w_kernel<<<dim3(1024), dim3(256), 0, stream>>>(wq, wk, wv, WQH, WKH, WVH);

  proj_gemm_kernel<<<dim3(64, 2, 2),  dim3(256), 0, stream>>>(WQH, XHT, bq, QT, 64, 1);
  proj_gemm_kernel<<<dim3(64, 2, 2),  dim3(256), 0, stream>>>(WKH, XHT, bk, KT, 64, 1);
  proj_gemm_kernel<<<dim3(64, 16, 2), dim3(256), 0, stream>>>(WVH, XHT, bv, VH, 512, 0);

  chan_energy_kernel<<<dim3(8, 4, 2), dim3(256), 0, stream>>>(XH, EC);
  chan_softmax_kernel<<<dim3(1024), dim3(256), 0, stream>>>(EC, ATTC);

  pos_flash_kernel<<<dim3(128, 2), dim3(256), (size_t)2 * VBUF * sizeof(_Float16), stream>>>(
      QT, KT, VH, POS);

  chan_out_combine_kernel<<<dim3(64, 16, 2), dim3(256), 0, stream>>>(
      ATTC, XHT, POS, x, gp, gc, al, be, out);
}
